// RSSMCore_76424648065481
// MI455X (gfx1250) — compile-verified, matrix operand order checked by reading, not run on hardware
//
#include <hip/hip_runtime.h>
#include <hip/hip_bf16.h>
#include <stdint.h>

// ---------------------------------------------------------------------------
// RSSM core for MI455X (gfx1250, wave32, WMMA).
// - one-time f32->bf16 transposed weight conversion (weights live in 192MB L2)
// - hoisted step-independent encoder GEMMs
// - per-step WMMA GEMMs (v_wmma_f32_16x16x32_bf16, f32 accumulate)
// - B-operand staged in LDS via async global->LDS loads (double-buffered),
//   shared by all 8 waves of the block (8x global-traffic reduction)
// - fused GRU gates, LayerNorm+ELU epilogues, Gumbel-argmax one-hot sampling
// - batched prior head at the end
// ---------------------------------------------------------------------------

typedef __bf16 bf16;
typedef __attribute__((ext_vector_type(16))) __bf16 v16bf;
typedef __attribute__((ext_vector_type(8)))  __bf16 v8bf;
typedef __attribute__((ext_vector_type(8)))  float  v8f;
typedef __attribute__((ext_vector_type(4)))  int    v4i;

#define O_STEPS 64
#define NB      512               // N*B = 8*64 rows per step
#define H_DIM   1024
#define D_DIM   2048
#define Z_DIM   1024
#define D3      6144
#define M_ALL   (O_STEPS * NB)    // 32768
#define KCHUNK  256               // K-chunk staged in LDS per buffer

// ---------------- async global->LDS copy (CDNA5 ASYNCcnt path) --------------
#if defined(__has_builtin)
#  if __has_builtin(__builtin_amdgcn_global_load_async_to_lds_b128) && \
      __has_builtin(__builtin_amdgcn_s_wait_asynccnt)
#    define USE_ASYNC_LDS 1
#  endif
#endif
#ifndef USE_ASYNC_LDS
#  define USE_ASYNC_LDS 0
#endif

#if USE_ASYNC_LDS
typedef __attribute__((address_space(1))) v4i as1_v4i;   // global (device)
typedef __attribute__((address_space(3))) v4i as3_v4i;   // LDS (shared)
#endif

__device__ __forceinline__ void cp_b16(const bf16* gp, bf16* lp) {
#if USE_ASYNC_LDS
    __builtin_amdgcn_global_load_async_to_lds_b128(
        (as1_v4i*)gp, (as3_v4i*)lp, /*offset=*/0, /*cpol=*/0);
#else
    *reinterpret_cast<v8bf*>(lp) = *reinterpret_cast<const v8bf*>(gp);
#endif
}
__device__ __forceinline__ void cp_wait() {
#if USE_ASYNC_LDS
    __builtin_amdgcn_s_wait_asynccnt(0);
#endif
}

// ---------------- fragment load (16-bit A/B layout, ISA 7.12.2) -------------
// lane L: matrix-row = L&15 ; K-chunks {koff..koff+7, koff+16..koff+23},
// koff = (L>>4)*8. Two contiguous 16B loads.
__device__ __forceinline__ v16bf ldfrag(const bf16* __restrict__ p) {
    v8bf lo = *reinterpret_cast<const v8bf*>(p);
    v8bf hi = *reinterpret_cast<const v8bf*>(p + 16);
    return __builtin_shufflevector(lo, hi, 0,1,2,3,4,5,6,7,8,9,10,11,12,13,14,15);
}

__device__ __forceinline__ v8f wmma_bf16(v16bf a, v16bf b, v8f c) {
    return __builtin_amdgcn_wmma_f32_16x16x32_bf16(
        false, a, false, b, (short)0, c, false, false);
}

__device__ __forceinline__ float eluf(float v)    { return v > 0.0f ? v : expm1f(v); }
__device__ __forceinline__ float sigmoidf(float v){ return 1.0f / (1.0f + expf(-v)); }

// ---------------- generic WMMA GEMM: C = A(MxK) * Bt(NxK)^T -----------------
// Block: 8 waves, 128 rows x 64 cols. B strip (shared by all waves) staged in
// LDS, double-buffered in KCHUNK slices via async loads.
// EPI 0: Cf = acc + bias ; EPI 1: += addend ; EPI 2: Cb = bf16(elu(acc+bias))
template <int EPI>
__global__ __launch_bounds__(256) void gemm_wmma(
    const bf16*  __restrict__ A,  int lda,
    const bf16*  __restrict__ Bt, int ldb,
    const float* __restrict__ bias,
    const float* __restrict__ addend,
    float* __restrict__ Cf, bf16* __restrict__ Cb,
    int ldc, int K)
{
    __shared__ __align__(16) bf16 sB[2][64 * KCHUNK];

    const int tid  = threadIdx.x;
    const int lane = tid & 31;
    const int wv   = tid >> 5;
    const int row0 = blockIdx.y * 128 + wv * 16;
    const int col0 = blockIdx.x * 64;
    const int lr   = lane & 15;
    const int koff = (lane >> 4) << 3;
    const int rb   = (lane >> 4) * 8;

    const bf16* ap = A + (size_t)(row0 + lr) * lda + koff;

    // cooperative stage of 64 x KCHUNK B slice at k-offset kc into buffer bi
    auto stage = [&](int bi, int kc) {
        const int ntr = 64 * KCHUNK / 8;               // 16B transfers
#pragma unroll
        for (int i = tid; i < ntr; i += 256) {
            const int r = i / (KCHUNK / 8);
            const int c = (i % (KCHUNK / 8)) * 8;
            cp_b16(Bt + (size_t)(col0 + r) * ldb + kc + c, &sB[bi][r * KCHUNK + c]);
        }
    };

    v8f acc[4] = {v8f{}, v8f{}, v8f{}, v8f{}};
    const int nch = K / KCHUNK;
    stage(0, 0);
    for (int c = 0; c < nch; ++c) {
        cp_wait();
        __syncthreads();                    // buf[c&1] ready; prev reads done
        if (c + 1 < nch) stage((c + 1) & 1, (c + 1) * KCHUNK);
        const bf16* bb = &sB[c & 1][0];
        for (int kk = 0; kk < KCHUNK; kk += 32) {
            __builtin_prefetch(ap + c * KCHUNK + kk + 256, 0, 1);
            v16bf a  = ldfrag(ap + c * KCHUNK + kk);
            v16bf b0 = ldfrag(bb + (size_t)(0 * 16 + lr) * KCHUNK + kk + koff);
            v16bf b1 = ldfrag(bb + (size_t)(1 * 16 + lr) * KCHUNK + kk + koff);
            v16bf b2 = ldfrag(bb + (size_t)(2 * 16 + lr) * KCHUNK + kk + koff);
            v16bf b3 = ldfrag(bb + (size_t)(3 * 16 + lr) * KCHUNK + kk + koff);
            acc[0] = wmma_bf16(a, b0, acc[0]);
            acc[1] = wmma_bf16(a, b1, acc[1]);
            acc[2] = wmma_bf16(a, b2, acc[2]);
            acc[3] = wmma_bf16(a, b3, acc[3]);
        }
    }
#pragma unroll
    for (int j = 0; j < 4; ++j) {
        const int n = col0 + j * 16 + lr;
        const float bv = bias ? bias[n] : 0.0f;
#pragma unroll
        for (int v = 0; v < 8; ++v) {
            const int m = row0 + rb + v;
            const size_t idx = (size_t)m * ldc + n;
            float val = acc[j][v] + bv;
            if constexpr (EPI == 1) val += addend[idx];
            if constexpr (EPI == 2) Cb[idx] = (bf16)eluf(val);
            else                    Cf[idx] = val;
        }
    }
}

// ---------------- fused GRU: 6 tiles per wave share A fragments -------------
// Block: 8 waves, 128 rows x one 16-wide d tile. The 3-gate B columns for both
// W_ih and W_hh are shared by all waves -> staged in LDS (48 rows x KCHUNK).
__global__ __launch_bounds__(256) void gru_wmma(
    const bf16*  __restrict__ za,    // 512 x 1024 bf16
    const bf16*  __restrict__ hmb,   // 512 x 2048 bf16 (masked h)
    const float* __restrict__ hmf,   // 512 x 2048 f32  (masked h)
    const bf16*  __restrict__ WihT,  // 6144 x 1024
    const bf16*  __restrict__ WhhT,  // 6144 x 2048
    const float* __restrict__ b_ih, const float* __restrict__ b_hh,
    float* __restrict__ h_state,     // 512 x 2048
    bf16*  __restrict__ h_all_b,     // this step's slice, 512 x 2048
    float* __restrict__ out_states,  // this step's slice, 512 x 2048
    float* __restrict__ out_feat)    // this step's slice, 512 x 3072 ([0,2048))
{
    __shared__ __align__(16) bf16 sB[2][48 * KCHUNK];

    const int tid  = threadIdx.x;
    const int lane = tid & 31;
    const int wv   = tid >> 5;
    const int row0 = blockIdx.y * 128 + wv * 16;
    const int d0   = blockIdx.x * 16;
    const int lr   = lane & 15;
    const int koff = (lane >> 4) << 3;
    const int rb   = (lane >> 4) * 8;

    v8f acc[6] = {v8f{}, v8f{}, v8f{}, v8f{}, v8f{}, v8f{}};

    // stage 3 gate row-blocks (16 rows each) of W at k-offset kc
    auto stage = [&](const bf16* W, int ldw, int bi, int kc) {
        const int ntr = 48 * KCHUNK / 8;
#pragma unroll
        for (int i = tid; i < ntr; i += 256) {
            const int r48 = i / (KCHUNK / 8);          // 0..47
            const int c   = (i % (KCHUNK / 8)) * 8;
            const int g   = r48 >> 4;
            const int rr  = r48 & 15;
            cp_b16(W + (size_t)(g * D_DIM + d0 + rr) * ldw + kc + c,
                   &sB[bi][r48 * KCHUNK + c]);
        }
    };

    // ---- pass over both GEMMs: (za @ W_ih, K=1024) then (hm @ W_hh, K=2048)
    for (int pass = 0; pass < 2; ++pass) {
        const bf16*  Wt  = pass ? WhhT : WihT;
        const bf16*  Ain = pass ? hmb  : za;
        const int    Kp  = pass ? D_DIM : H_DIM;
        const int    aoff = pass ? 3 : 0;
        const bf16* ap = Ain + (size_t)(row0 + lr) * Kp + koff;

        const int nch = Kp / KCHUNK;
        stage(Wt, Kp, 0, 0);
        for (int c = 0; c < nch; ++c) {
            cp_wait();
            __syncthreads();
            if (c + 1 < nch) stage(Wt, Kp, (c + 1) & 1, (c + 1) * KCHUNK);
            const bf16* bb = &sB[c & 1][0];
            for (int kk = 0; kk < KCHUNK; kk += 32) {
                __builtin_prefetch(ap + c * KCHUNK + kk + 256, 0, 1);
                v16bf a = ldfrag(ap + c * KCHUNK + kk);
#pragma unroll
                for (int g = 0; g < 3; ++g) {
                    v16bf b = ldfrag(bb + (size_t)(g * 16 + lr) * KCHUNK + kk + koff);
                    acc[aoff + g] = wmma_bf16(a, b, acc[aoff + g]);
                }
            }
        }
        cp_wait();
        __syncthreads();    // drain before re-staging buffers for next pass
    }

    const int d = d0 + lr;
    const float bir = b_ih[d], biu = b_ih[D_DIM + d], bin = b_ih[2 * D_DIM + d];
    const float bhr = b_hh[d], bhu = b_hh[D_DIM + d], bhn = b_hh[2 * D_DIM + d];
#pragma unroll
    for (int v = 0; v < 8; ++v) {
        const int m = row0 + rb + v;
        const float r  = sigmoidf(acc[0][v] + bir + acc[3][v] + bhr);
        const float u  = sigmoidf(acc[1][v] + biu + acc[4][v] + bhu);
        const float nn = tanhf(acc[2][v] + bin + r * (acc[5][v] + bhn));
        const float hold = hmf[(size_t)m * D_DIM + d];
        const float hn = (1.0f - u) * nn + u * hold;
        const size_t idx = (size_t)m * D_DIM + d;
        h_state[idx]    = hn;
        h_all_b[idx]    = (bf16)hn;
        out_states[idx] = hn;
        out_feat[(size_t)m * 3072 + d] = hn;
    }
}

// ---------------- LayerNorm + ELU -> bf16 (one block per row) ---------------
__global__ __launch_bounds__(256) void ln_elu_bf16(
    const float* __restrict__ X, const float* __restrict__ g,
    const float* __restrict__ b, bf16* __restrict__ Y, int Hd)
{
    const int row = blockIdx.x;
    const float* x = X + (size_t)row * Hd;
    __shared__ float red[2][8];
    float s = 0.0f, ss = 0.0f;
    for (int c = threadIdx.x; c < Hd; c += 256) {
        const float v = x[c]; s += v; ss += v * v;
    }
    for (int o = 16; o > 0; o >>= 1) { s += __shfl_xor(s, o, 32); ss += __shfl_xor(ss, o, 32); }
    if ((threadIdx.x & 31) == 0) { red[0][threadIdx.x >> 5] = s; red[1][threadIdx.x >> 5] = ss; }
    __syncthreads();
    if (threadIdx.x < 32) {
        float a  = (threadIdx.x < 8) ? red[0][threadIdx.x] : 0.0f;
        float a2 = (threadIdx.x < 8) ? red[1][threadIdx.x] : 0.0f;
        for (int o = 4; o > 0; o >>= 1) { a += __shfl_xor(a, o, 32); a2 += __shfl_xor(a2, o, 32); }
        if (threadIdx.x == 0) { red[0][0] = a; red[1][0] = a2; }
    }
    __syncthreads();
    const float mean = red[0][0] / (float)Hd;
    const float var  = red[1][0] / (float)Hd - mean * mean;
    const float inv  = rsqrtf(var + 1e-3f);
    for (int c = threadIdx.x; c < Hd; c += 256) {
        float v = (x[c] - mean) * inv * g[c] + b[c];
        Y[(size_t)row * Hd + c] = (bf16)eluf(v);
    }
}

// ---------------- weight transpose + cast: W(KxN) f32 -> Wt(NxK) bf16 -------
__global__ __launch_bounds__(256) void transpose_cast_bf16(
    const float* __restrict__ W, bf16* __restrict__ Wt, int K, int N)
{
    __shared__ float tile[32][33];
    const int nb = blockIdx.x * 32, kb = blockIdx.y * 32;
    const int tx = threadIdx.x & 31, ty = threadIdx.x >> 5;  // (32,8)
#pragma unroll
    for (int i = 0; i < 32; i += 8)
        tile[ty + i][tx] = W[(size_t)(kb + ty + i) * N + (nb + tx)];
    __syncthreads();
#pragma unroll
    for (int i = 0; i < 32; i += 8)
        Wt[(size_t)(nb + ty + i) * K + (kb + tx)] = (bf16)tile[tx][ty + i];
}

__global__ __launch_bounds__(256) void cast_bf16_kernel(
    const float* __restrict__ X, bf16* __restrict__ Y, size_t n)
{
    for (size_t i = (size_t)blockIdx.x * 256 + threadIdx.x; i < n;
         i += (size_t)gridDim.x * 256)
        Y[i] = (bf16)X[i];
}

// ---------------- mask: m[b] = all_n(!reset[t,n,b]) -------------------------
__global__ void compute_mask_kernel(const unsigned char* __restrict__ reset_t,
                                    float* __restrict__ mask)
{
    const int b = threadIdx.x;            // 64 threads
    unsigned any = 0;
#pragma unroll
    for (int n = 0; n < 8; ++n) any |= reset_t[n * 64 + b];
    mask[b] = any ? 0.0f : 1.0f;
}

__global__ __launch_bounds__(256) void apply_mask_h_kernel(
    const float* __restrict__ h, const float* __restrict__ mask,
    float* __restrict__ hm_f, bf16* __restrict__ hm_b)
{
    const size_t i = (size_t)blockIdx.x * 256 + threadIdx.x;   // 512*2048 total
    const int b = (int)((i >> 11) & 63);
    const float v = h[i] * mask[b];
    hm_f[i] = v; hm_b[i] = (bf16)v;
}

__global__ __launch_bounds__(256) void apply_mask_z_kernel(
    const float* __restrict__ z, const float* __restrict__ mask,
    bf16* __restrict__ zm_b)
{
    const size_t i = (size_t)blockIdx.x * 256 + threadIdx.x;   // 512*1024 total
    const int b = (int)((i >> 10) & 63);
    zm_b[i] = (bf16)(z[i] * mask[b]);
}

// ---------------- Gumbel-argmax one-hot sample ------------------------------
// Straight-through value: onehot + probs - probs  ==  onehot.
__device__ __forceinline__ uint32_t pcg_hash(uint32_t v) {
    v = v * 747796405u + 2891336453u;
    uint32_t w = ((v >> ((v >> 28u) + 4u)) ^ v) * 277803737u;
    return (w >> 22u) ^ w;
}

__global__ __launch_bounds__(256) void sample_kernel(
    const float* __restrict__ posts_t,   // this step, 512 x 1024
    float* __restrict__ out_s1, float* __restrict__ out_s2,
    float* __restrict__ out_feat,        // this step, 512 x 3072 ([2048,3072))
    float* __restrict__ z_state, int t)
{
    const int gw   = (int)((blockIdx.x * 256 + threadIdx.x) >> 5); // 0..16383
    const int lane = threadIdx.x & 31;
    const int row  = gw >> 5;   // 512 rows
    const int s    = gw & 31;   // 32 stoch dims
    const size_t col = (size_t)s * 32 + lane;
    const float logit = posts_t[(size_t)row * Z_DIM + col];

    const uint32_t seed = ((uint32_t)t * 0x9E3779B9u) ^ ((uint32_t)row << 10) ^
                          ((uint32_t)s << 5) ^ (uint32_t)lane;
    const float u = (float)(pcg_hash(seed) >> 8) * (1.0f / 16777216.0f) + 1e-7f;
    float key = logit + (-logf(-logf(u)));
    int   idx = lane;
    for (int o = 16; o > 0; o >>= 1) {
        const float ok = __shfl_xor(key, o, 32);
        const int   oi = __shfl_xor(idx, o, 32);
        if (ok > key || (ok == key && oi < idx)) { key = ok; idx = oi; }
    }
    const float one = (lane == idx) ? 1.0f : 0.0f;
    out_s1[(size_t)row * Z_DIM + col] = one;
    out_s2[(size_t)row * Z_DIM + col] = one;
    out_feat[(size_t)row * 3072 + D_DIM + col] = one;
    z_state[(size_t)row * Z_DIM + col] = one;
}

// ===========================================================================
extern "C" void kernel_launch(void* const* d_in, const int* in_sizes, int n_in,
                              void* d_out, int out_size, void* d_ws, size_t ws_size,
                              hipStream_t stream)
{
    (void)in_sizes; (void)n_in; (void)out_size; (void)ws_size;
    // ------------- inputs (setup_inputs dict order) -------------
    const float*         embed = (const float*)d_in[0];
    const unsigned char* reset = (const unsigned char*)d_in[1];
    const float* in_h  = (const float*)d_in[2];
    const float* in_z  = (const float*)d_in[3];
    const float* W_emb = (const float*)d_in[4];
    const float* b_emb = (const float*)d_in[5];
    const float* W_z   = (const float*)d_in[6];
    const float* b_z   = (const float*)d_in[7];
    const float* g_in  = (const float*)d_in[8];
    const float* b_in  = (const float*)d_in[9];
    const float* W_ih  = (const float*)d_in[10];
    const float* b_ih  = (const float*)d_in[11];
    const float* W_hh  = (const float*)d_in[12];
    const float* b_hh  = (const float*)d_in[13];
    const float* W_prh = (const float*)d_in[14];
    const float* b_prh = (const float*)d_in[15];
    const float* g_pr  = (const float*)d_in[16];
    const float* b_pr  = (const float*)d_in[17];
    const float* W_pr  = (const float*)d_in[18];
    const float* b_pro = (const float*)d_in[19];
    const float* W_poh = (const float*)d_in[20];
    const float* b_poh = (const float*)d_in[21];
    const float* W_poe = (const float*)d_in[22];
    const float* g_po  = (const float*)d_in[23];
    const float* b_po  = (const float*)d_in[24];
    const float* W_po  = (const float*)d_in[25];
    const float* b_poo = (const float*)d_in[26];

    float* out = (float*)d_out;
    // ------------- output offsets (flat f32 elements, return order) -------------
    const size_t o_priors   = 0;
    const size_t o_posts    = 33554432;
    const size_t o_samples  = 67108864;
    const size_t o_features = 100663296;
    const size_t o_states   = 201326592;
    const size_t o_samples2 = 268435456;
    const size_t o_hlast    = 301989888;
    const size_t o_zlast    = 303038464;

    // ------------- workspace layout -------------
    char* ws = (char*)d_ws;
    size_t off = 0;
    auto alloc = [&](size_t bytes) { size_t o = off; off += (bytes + 255) & ~(size_t)255; return ws + o; };
    bf16* WembT = (bf16*)alloc((size_t)H_DIM * H_DIM * 2);
    bf16* WzT   = (bf16*)alloc((size_t)H_DIM * H_DIM * 2);
    bf16* WihT  = (bf16*)alloc((size_t)D3 * H_DIM * 2);
    bf16* WhhT  = (bf16*)alloc((size_t)D3 * D_DIM * 2);
    bf16* WpohT = (bf16*)alloc((size_t)H_DIM * D_DIM * 2);
    bf16* WpoeT = (bf16*)alloc((size_t)H_DIM * H_DIM * 2);
    bf16* WpoT  = (bf16*)alloc((size_t)Z_DIM * H_DIM * 2);
    bf16* WprhT = (bf16*)alloc((size_t)H_DIM * D_DIM * 2);
    bf16* WprT  = (bf16*)alloc((size_t)Z_DIM * H_DIM * 2);
    bf16*  emb_b   = (bf16*)alloc((size_t)M_ALL * H_DIM * 2);
    bf16*  E_b     = (bf16*)alloc((size_t)M_ALL * H_DIM * 2);  // reused: xp bf16 (prior head)
    float* EPOE    = (float*)alloc((size_t)M_ALL * H_DIM * 4); // reused: prior pre-LN
    bf16*  h_all_b = (bf16*)alloc((size_t)M_ALL * D_DIM * 2);
    float* h_state = (float*)alloc((size_t)NB * D_DIM * 4);
    float* z_state = (float*)alloc((size_t)NB * Z_DIM * 4);
    float* hm_f    = (float*)alloc((size_t)NB * D_DIM * 4);
    bf16*  hm_b    = (bf16*)alloc((size_t)NB * D_DIM * 2);
    bf16*  zm_b    = (bf16*)alloc((size_t)NB * Z_DIM * 2);
    float* t0      = (float*)alloc((size_t)NB * H_DIM * 4);
    bf16*  za_b    = (bf16*)alloc((size_t)NB * H_DIM * 2);
    float* xpre    = (float*)alloc((size_t)NB * H_DIM * 4);
    bf16*  xact_b  = (bf16*)alloc((size_t)NB * H_DIM * 2);
    float* maskbuf = (float*)alloc(64 * 4);

    const dim3 blk(256);
    const dim3 t32(32 * 8);

    // ------------- one-time weight conversion (bf16, transposed NxK) -------------
    transpose_cast_bf16<<<dim3(H_DIM/32, H_DIM/32), t32, 0, stream>>>(W_emb, WembT, H_DIM, H_DIM);
    transpose_cast_bf16<<<dim3(H_DIM/32, H_DIM/32), t32, 0, stream>>>(W_z,   WzT,   H_DIM, H_DIM);
    transpose_cast_bf16<<<dim3(D3/32,    H_DIM/32), t32, 0, stream>>>(W_ih,  WihT,  H_DIM, D3);
    transpose_cast_bf16<<<dim3(D3/32,    D_DIM/32), t32, 0, stream>>>(W_hh,  WhhT,  D_DIM, D3);
    transpose_cast_bf16<<<dim3(H_DIM/32, D_DIM/32), t32, 0, stream>>>(W_poh, WpohT, D_DIM, H_DIM);
    transpose_cast_bf16<<<dim3(H_DIM/32, H_DIM/32), t32, 0, stream>>>(W_poe, WpoeT, H_DIM, H_DIM);
    transpose_cast_bf16<<<dim3(Z_DIM/32, H_DIM/32), t32, 0, stream>>>(W_po,  WpoT,  H_DIM, Z_DIM);
    transpose_cast_bf16<<<dim3(H_DIM/32, D_DIM/32), t32, 0, stream>>>(W_prh, WprhT, D_DIM, H_DIM);
    transpose_cast_bf16<<<dim3(Z_DIM/32, H_DIM/32), t32, 0, stream>>>(W_pr,  WprT,  H_DIM, Z_DIM);

    // ------------- hoisted encoder: E = elu(embed@W_emb+b), EPOE = E@W_poe ------
    cast_bf16_kernel<<<4096, blk, 0, stream>>>(embed, emb_b, (size_t)M_ALL * H_DIM);
    gemm_wmma<2><<<dim3(H_DIM/64, M_ALL/128), blk, 0, stream>>>(
        emb_b, H_DIM, WembT, H_DIM, b_emb, nullptr, nullptr, E_b, H_DIM, H_DIM);
    gemm_wmma<0><<<dim3(H_DIM/64, M_ALL/128), blk, 0, stream>>>(
        E_b, H_DIM, WpoeT, H_DIM, nullptr, nullptr, EPOE, nullptr, H_DIM, H_DIM);

    // ------------- initial state -------------
    (void)hipMemcpyAsync(h_state, in_h, (size_t)NB * D_DIM * 4, hipMemcpyDeviceToDevice, stream);
    (void)hipMemcpyAsync(z_state, in_z, (size_t)NB * Z_DIM * 4, hipMemcpyDeviceToDevice, stream);

    // ------------- sequential scan over O steps -------------
    for (int t = 0; t < O_STEPS; ++t) {
        const size_t step = (size_t)t * NB;
        // mask
        compute_mask_kernel<<<1, 64, 0, stream>>>(reset + step, maskbuf);
        apply_mask_h_kernel<<<(NB * D_DIM) / 256, blk, 0, stream>>>(h_state, maskbuf, hm_f, hm_b);
        apply_mask_z_kernel<<<(NB * Z_DIM) / 256, blk, 0, stream>>>(z_state, maskbuf, zm_b);
        // za = elu(LN(zm @ W_z + b_z) * g_in + b_in)
        gemm_wmma<0><<<dim3(H_DIM/64, NB/128), blk, 0, stream>>>(
            zm_b, Z_DIM, WzT, Z_DIM, b_z, nullptr, t0, nullptr, H_DIM, Z_DIM);
        ln_elu_bf16<<<NB, blk, 0, stream>>>(t0, g_in, b_in, za_b, H_DIM);
        // GRU: h_new (also fans out to states_h / features / bf16 copy)
        gru_wmma<<<dim3(D_DIM/16, NB/128), blk, 0, stream>>>(
            za_b, hm_b, hm_f, WihT, WhhT, b_ih, b_hh,
            h_state, h_all_b + step * D_DIM,
            out + o_states + step * D_DIM,
            out + o_features + step * 3072);
        // x = LN(h @ W_poh + b_poh + EPOE) ; post = elu(x) @ W_po + b_poo
        gemm_wmma<1><<<dim3(H_DIM/64, NB/128), blk, 0, stream>>>(
            h_all_b + step * D_DIM, D_DIM, WpohT, D_DIM, b_poh,
            EPOE + step * H_DIM, xpre, nullptr, H_DIM, D_DIM);
        ln_elu_bf16<<<NB, blk, 0, stream>>>(xpre, g_po, b_po, xact_b, H_DIM);
        gemm_wmma<0><<<dim3(Z_DIM/64, NB/128), blk, 0, stream>>>(
            xact_b, H_DIM, WpoT, H_DIM, b_poo, nullptr,
            out + o_posts + step * Z_DIM, nullptr, Z_DIM, H_DIM);
        // straight-through Gumbel one-hot sample -> z_new
        sample_kernel<<<(NB * 32 * 32) / 256, blk, 0, stream>>>(
            out + o_posts + step * Z_DIM,
            out + o_samples + step * Z_DIM,
            out + o_samples2 + step * Z_DIM,
            out + o_features + step * 3072,
            z_state, t);
    }

    // ------------- batched prior head over all states_h -------------
    gemm_wmma<0><<<dim3(H_DIM/64, M_ALL/128), blk, 0, stream>>>(
        h_all_b, D_DIM, WprhT, D_DIM, b_prh, nullptr, EPOE, nullptr, H_DIM, D_DIM);
    ln_elu_bf16<<<M_ALL, blk, 0, stream>>>(EPOE, g_pr, b_pr, E_b, H_DIM);
    gemm_wmma<0><<<dim3(Z_DIM/64, M_ALL/128), blk, 0, stream>>>(
        E_b, H_DIM, WprT, H_DIM, b_pro, nullptr, out + o_priors, nullptr, Z_DIM, H_DIM);

    // ------------- h_last / z_last -------------
    (void)hipMemcpyAsync(out + o_hlast, h_state, (size_t)NB * D_DIM * 4, hipMemcpyDeviceToDevice, stream);
    (void)hipMemcpyAsync(out + o_zlast, z_state, (size_t)NB * Z_DIM * 4, hipMemcpyDeviceToDevice, stream);
}
